// ContextPooling_3049426780885
// MI455X (gfx1250) — compile-verified, matrix operand order checked by reading, not run on hardware
//
#include <hip/hip_runtime.h>
#include <hip/hip_bf16.h>

#define DEVINL __device__ __forceinline__

typedef __attribute__((ext_vector_type(16))) __bf16 v16bf;
typedef __attribute__((ext_vector_type(8)))  float  v8f;
typedef __attribute__((ext_vector_type(8)))  unsigned short u16x8;

constexpr int B_  = 8;
constexpr int L_  = 1024;
constexpr int D_  = 1024;
constexpr int C1_ = 2048;
constexpr float SLOPE_   = 0.01f;
constexpr float EPS_LN_  = 1e-5f;
constexpr float EPS_STD_ = 1e-5f;
constexpr float R_       = 0.1f;

DEVINL unsigned short f2bf(float f) {
  unsigned u = __float_as_uint(f);
  u += 0x7FFFu + ((u >> 16) & 1u);   // round-to-nearest-even
  return (unsigned short)(u >> 16);
}
DEVINL float bf2f(unsigned short h) {
  return __uint_as_float(((unsigned)h) << 16);
}
DEVINL float lrelu(float v) { return v > 0.0f ? v : SLOPE_ * v; }

// CDNA5 async copy global->LDS (ASYNCcnt), 16B per active lane.
DEVINL void async_b128(const unsigned short* lds_dst,
                       const unsigned short* gbase, unsigned byte_off) {
  unsigned l = (unsigned)(size_t)lds_dst;        // flat shared addr low 32 = LDS offset
  asm volatile("global_load_async_to_lds_b128 %0, %1, %2"
               :: "v"(l), "v"(byte_off),
                  "s"((unsigned long long)(size_t)gbase)
               : "memory");
}
DEVINL void async_wait0() {
  asm volatile("s_wait_asynccnt 0" ::: "memory");
}

// ---------------------------------------------------------------------------
// Pack conv weight [Cout][Cin][3] (f32) -> [3][Cout][Cin] (bf16)
// ---------------------------------------------------------------------------
__global__ void pack_w_kernel(const float* __restrict__ w,
                              unsigned short* __restrict__ wp,
                              int Cout, int Cin) {
  const size_t total = (size_t)3 * Cout * Cin;
  const size_t n = (size_t)Cout * Cin;
  for (size_t idx = (size_t)blockIdx.x * blockDim.x + threadIdx.x;
       idx < total; idx += (size_t)gridDim.x * blockDim.x) {
    int t  = (int)(idx / n);
    size_t r = idx % n;
    int co = (int)(r / Cin);
    int ci = (int)(r % Cin);
    wp[idx] = f2bf(w[((size_t)co * Cin + ci) * 3 + t]);
  }
}

// ---------------------------------------------------------------------------
// feats [B][L][D] f32 -> xT [B][D][L] bf16 (LDS tile transpose)
//                     -> fbf [B][L][D] bf16 (straight copy)
// fbf is conv1's transposed-B; xT is the final GEMM's transposed-B.
// ---------------------------------------------------------------------------
__global__ void transpose_convert_kernel(const float* __restrict__ feats,
                                         unsigned short* __restrict__ xT,
                                         unsigned short* __restrict__ fbf) {
  __shared__ unsigned short tile[32][33];
  const int b  = blockIdx.z;
  const int d0 = blockIdx.x * 32;
  const int l0 = blockIdx.y * 32;
  #pragma unroll
  for (int i = 0; i < 4; ++i) {
    int l = l0 + threadIdx.y + i * 8;
    int d = d0 + threadIdx.x;
    size_t gi = ((size_t)b * L_ + l) * D_ + d;
    unsigned short h = f2bf(feats[gi]);
    fbf[gi] = h;
    tile[threadIdx.y + i * 8][threadIdx.x] = h;
  }
  __syncthreads();
  #pragma unroll
  for (int i = 0; i < 4; ++i) {
    int d = d0 + threadIdx.y + i * 8;
    int l = l0 + threadIdx.x;
    xT[((size_t)b * D_ + d) * L_ + l] = tile[threadIdx.x][threadIdx.y + i * 8];
  }
}

// ---------------------------------------------------------------------------
// bf16 WMMA GEMM with optional 3-tap conv accumulation, async double-buffered.
//   C[b][m][n] = sum_t sum_k A[t][m][k] * BT[b][n + t - HALO][k]  (+ bias[m])
// BT is the B matrix stored transposed: BT[n][k], row stride K.
// Block tile 128x128, K-step 32, 256 threads = 8 waves, each wave 32x64.
// B tile is staged once per K-step with a +-1 column halo; taps just shift the
// LDS row index. All staging is contiguous b128 -> async-to-LDS engine.
// ---------------------------------------------------------------------------
template<int TAPS, bool HAS_BIAS>
__launch_bounds__(256)
__global__ void gemm_wmma_kernel(const unsigned short* __restrict__ A,
                                 unsigned a_tap_stride_el, size_t a_batch_stride,
                                 const unsigned short* __restrict__ BT,
                                 const float* __restrict__ bias,
                                 float* __restrict__ C,
                                 int M, int N, int K) {
  constexpr int HALO  = (TAPS > 1) ? 1 : 0;
  constexpr int BROWS = 128 + 2 * HALO;
  __shared__ unsigned short lA[2][TAPS][128][40];  // 80B row stride, 16B aligned
  __shared__ unsigned short lB[2][BROWS][40];

  const int tid = threadIdx.x;
  const int b   = blockIdx.z;
  const int m0  = blockIdx.y * 128;
  const int n0  = blockIdx.x * 128;

  const unsigned short* Bb = BT + (size_t)b * (size_t)N * K;
  const unsigned short* Ab = A  + (size_t)b * a_batch_stride;
  float* Cb = C + (size_t)b * (size_t)M * N;

  const int wave = tid >> 5;
  const int lane = tid & 31;
  const int wm   = wave >> 1;     // 0..3 (M slice of 32)
  const int wn   = wave & 1;      // 0..1 (N slice of 64)
  const int half = lane >> 4;     // K-group select
  const int lrow = lane & 15;     // row/col within 16

  const int r2 = tid >> 1;        // staging row 0..127
  const int c2 = (tid & 1) * 16;  // staging col {0,16}

  v8f acc[2][4] = {};
  const int ksteps = K >> 5;

  auto stage = [&](int kb, int buf) {
    const unsigned kbase = (unsigned)(kb << 5);
    // A tiles, all taps
    #pragma unroll
    for (int t = 0; t < TAPS; ++t) {
      unsigned off = ((unsigned)t * a_tap_stride_el
                    + (unsigned)(m0 + r2) * (unsigned)K + kbase + (unsigned)c2) * 2u;
      async_b128(&lA[buf][t][r2][c2],     Ab, off);
      async_b128(&lA[buf][t][r2][c2 + 8], Ab, off + 16u);
    }
    // B main 128 rows (columns n0..n0+127)
    {
      unsigned off = ((unsigned)(n0 + r2) * (unsigned)K + kbase + (unsigned)c2) * 2u;
      async_b128(&lB[buf][HALO + r2][c2],     Bb, off);
      async_b128(&lB[buf][HALO + r2][c2 + 8], Bb, off + 16u);
    }
    // halo columns n0-1 and n0+128 (uniform branches; zeros at batch edges)
    if constexpr (HALO) {
      if (tid < 4) {
        if (n0 > 0) {
          unsigned off = ((unsigned)(n0 - 1) * (unsigned)K + kbase + (unsigned)tid * 8u) * 2u;
          async_b128(&lB[buf][0][tid * 8], Bb, off);
        } else {
          u16x8 z = {};
          *(u16x8*)&lB[buf][0][tid * 8] = z;
        }
      } else if (tid < 8) {
        int t2 = tid - 4;
        if (n0 + 128 < N) {
          unsigned off = ((unsigned)(n0 + 128) * (unsigned)K + kbase + (unsigned)t2 * 8u) * 2u;
          async_b128(&lB[buf][HALO + 128][t2 * 8], Bb, off);
        } else {
          u16x8 z = {};
          *(u16x8*)&lB[buf][HALO + 128][t2 * 8] = z;
        }
      }
    }
  };

  stage(0, 0);
  for (int kb = 0; kb < ksteps; ++kb) {
    const int cur = kb & 1;
    async_wait0();          // my async transfers for buf[cur] done
    __syncthreads();        // everyone's done; prior reads of buf[cur^1] retired
    if (kb + 1 < ksteps) stage(kb + 1, cur ^ 1);

    #pragma unroll
    for (int t = 0; t < TAPS; ++t) {
      v16bf af[2], bfm[4];
      #pragma unroll
      for (int mi = 0; mi < 2; ++mi) {
        const int row = wm * 32 + mi * 16 + lrow;
        union { u16x8 h[2]; v16bf v; } ua;
        ua.h[0] = *(const u16x8*)&lA[cur][t][row][half * 8];
        ua.h[1] = *(const u16x8*)&lA[cur][t][row][16 + half * 8];
        af[mi] = ua.v;
      }
      #pragma unroll
      for (int ni = 0; ni < 4; ++ni) {
        const int brow = wn * 64 + ni * 16 + lrow + t;   // tap shift via halo
        union { u16x8 h[2]; v16bf v; } ub;
        ub.h[0] = *(const u16x8*)&lB[cur][brow][half * 8];
        ub.h[1] = *(const u16x8*)&lB[cur][brow][16 + half * 8];
        bfm[ni] = ub.v;
      }
      #pragma unroll
      for (int mi = 0; mi < 2; ++mi)
        #pragma unroll
        for (int ni = 0; ni < 4; ++ni)
          acc[mi][ni] = __builtin_amdgcn_wmma_f32_16x16x32_bf16(
              false, af[mi], false, bfm[ni], (short)0, acc[mi][ni], false, false);
    }
  }

  // epilogue: C/D layout VGPR r -> M = r + 8*half, N = lrow
  #pragma unroll
  for (int mi = 0; mi < 2; ++mi) {
    #pragma unroll
    for (int ni = 0; ni < 4; ++ni) {
      const int colg = n0 + wn * 64 + ni * 16 + lrow;
      #pragma unroll
      for (int r = 0; r < 8; ++r) {
        const int rowg = m0 + wm * 32 + mi * 16 + half * 8 + r;
        float v = acc[mi][ni][r];
        if constexpr (HAS_BIAS) v += bias[rowg];
        Cb[(size_t)rowg * N + colg] = v;
      }
    }
  }
}

// ---------------------------------------------------------------------------
// LayerNorm over dims (C,L) per batch — two-stage deterministic reduction.
// ---------------------------------------------------------------------------
__global__ void ln_partial_kernel(const float* __restrict__ y,
                                  float2* __restrict__ part,
                                  int perBatch, int NB) {
  __shared__ float sred[256];
  __shared__ float qred[256];
  const int chunk = perBatch / NB;
  const float* yb = y + (size_t)blockIdx.y * perBatch + (size_t)blockIdx.x * chunk;
  float s = 0.0f, q = 0.0f;
  for (int i = threadIdx.x; i < chunk; i += 256) {
    float v = yb[i];
    s += v; q += v * v;
  }
  sred[threadIdx.x] = s; qred[threadIdx.x] = q;
  __syncthreads();
  for (int st = 128; st > 0; st >>= 1) {
    if (threadIdx.x < st) {
      sred[threadIdx.x] += sred[threadIdx.x + st];
      qred[threadIdx.x] += qred[threadIdx.x + st];
    }
    __syncthreads();
  }
  if (threadIdx.x == 0)
    part[(size_t)blockIdx.y * NB + blockIdx.x] = make_float2(sred[0], qred[0]);
}

__global__ void ln_finalize_kernel(const float2* __restrict__ part,
                                   float2* __restrict__ stats,
                                   int NB, float invN) {
  __shared__ float sred[256];
  __shared__ float qred[256];
  const int b = blockIdx.x;
  float s = 0.0f, q = 0.0f;
  for (int i = threadIdx.x; i < NB; i += 256) {
    float2 p = part[(size_t)b * NB + i];
    s += p.x; q += p.y;
  }
  sred[threadIdx.x] = s; qred[threadIdx.x] = q;
  __syncthreads();
  for (int st = 128; st > 0; st >>= 1) {
    if (threadIdx.x < st) {
      sred[threadIdx.x] += sred[threadIdx.x + st];
      qred[threadIdx.x] += qred[threadIdx.x + st];
    }
    __syncthreads();
  }
  if (threadIdx.x == 0) {
    float mean = sred[0] * invN;
    float var  = qred[0] * invN - mean * mean;
    stats[b] = make_float2(mean, rsqrtf(var + EPS_LN_));
  }
}

// LN apply + LeakyReLU + bf16 + transpose: y [B][C1][L] -> actT [B][L][C1]
__global__ void ln_apply_transpose_kernel(const float* __restrict__ y,
                                          const float2* __restrict__ stats,
                                          unsigned short* __restrict__ actT) {
  __shared__ unsigned short tile[32][33];
  const int b = blockIdx.z;
  const float2 st = stats[b];
  const int l0 = blockIdx.x * 32;
  const int c0 = blockIdx.y * 32;
  #pragma unroll
  for (int i = 0; i < 4; ++i) {
    int c = c0 + threadIdx.y + i * 8;
    int l = l0 + threadIdx.x;
    float v = (y[((size_t)b * C1_ + c) * L_ + l] - st.x) * st.y;
    tile[threadIdx.y + i * 8][threadIdx.x] = f2bf(lrelu(v));
  }
  __syncthreads();
  #pragma unroll
  for (int i = 0; i < 4; ++i) {
    int l = l0 + threadIdx.y + i * 8;
    int c = c0 + threadIdx.x;
    actT[((size_t)b * L_ + l) * C1_ + c] = tile[threadIdx.x][threadIdx.y + i * 8];
  }
}

// ---------------------------------------------------------------------------
// Head: conv3 (Cout=2, fp32) + LN over (2,L) + LeakyReLU + two softmaxes.
// actT layout [B][L][C1]; taps = clamped rows weighted by validity flags.
// ---------------------------------------------------------------------------
__launch_bounds__(1024)
__global__ void head_kernel(const unsigned short* __restrict__ actT, // [B][L][C1]
                            const float* __restrict__ w3,            // [2][C1][3]
                            const float* __restrict__ b3,            // [2]
                            float* __restrict__ s_out,               // [B][L]
                            float* __restrict__ w_out) {             // [B][L]
  __shared__ float red[1024];
  const int b = blockIdx.x;
  const int l = threadIdx.x;

  auto reduceSum = [&](float v) -> float {
    red[l] = v; __syncthreads();
    for (int st = 512; st > 0; st >>= 1) {
      if (l < st) red[l] += red[l + st];
      __syncthreads();
    }
    float r = red[0]; __syncthreads();
    return r;
  };
  auto reduceMax = [&](float v) -> float {
    red[l] = v; __syncthreads();
    for (int st = 512; st > 0; st >>= 1) {
      if (l < st) red[l] = fmaxf(red[l], red[l + st]);
      __syncthreads();
    }
    float r = red[0]; __syncthreads();
    return r;
  };

  const int lm = (l > 0) ? l - 1 : 0;
  const int lp = (l < L_ - 1) ? l + 1 : l;
  const float fm = (l > 0) ? 1.0f : 0.0f;
  const float fp = (l < L_ - 1) ? 1.0f : 0.0f;
  const unsigned short* rm = actT + ((size_t)b * L_ + lm) * C1_;
  const unsigned short* rc = actT + ((size_t)b * L_ + l ) * C1_;
  const unsigned short* rp = actT + ((size_t)b * L_ + lp) * C1_;

  float p00 = 0, p01 = 0, p02 = 0;   // ch0: taps (l-1, l, l+1)
  float p10 = 0, p11 = 0, p12 = 0;   // ch1
  for (int ci = 0; ci < C1_; ci += 8) {
    u16x8 vm = *(const u16x8*)(rm + ci);
    u16x8 vc = *(const u16x8*)(rc + ci);
    u16x8 vp = *(const u16x8*)(rp + ci);
    #pragma unroll
    for (int e = 0; e < 8; ++e) {
      float xm = bf2f(vm[e]), xc = bf2f(vc[e]), xp = bf2f(vp[e]);
      const float* w0 = w3 + (size_t)(ci + e) * 3;
      const float* w1 = w3 + ((size_t)C1_ + ci + e) * 3;
      p00 = fmaf(w0[0], xm, p00); p01 = fmaf(w0[1], xc, p01); p02 = fmaf(w0[2], xp, p02);
      p10 = fmaf(w1[0], xm, p10); p11 = fmaf(w1[1], xc, p11); p12 = fmaf(w1[2], xp, p12);
    }
  }
  float acc0 = b3[0] + fm * p00 + p01 + fp * p02;
  float acc1 = b3[1] + fm * p10 + p11 + fp * p12;

  // LayerNorm over all 2*L values of this batch
  float S = reduceSum(acc0 + acc1);
  float Q = reduceSum(acc0 * acc0 + acc1 * acc1);
  float mean = S / (2.0f * L_);
  float var  = Q / (2.0f * L_) - mean * mean;
  float inv  = rsqrtf(var + EPS_LN_);
  float a0 = lrelu((acc0 - mean) * inv);
  float a1 = lrelu((acc1 - mean) * inv);

  // softmax channel 0 -> s, channel 1 -> w
  float m0 = reduceMax(a0);
  float e0 = __expf(a0 - m0);
  float z0 = reduceSum(e0);
  s_out[(size_t)b * L_ + l] = e0 / z0;

  float m1 = reduceMax(a1);
  float e1 = __expf(a1 - m1);
  float z1 = reduceSum(e1);
  w_out[(size_t)b * L_ + l] = e1 / z1;
}

// ---------------------------------------------------------------------------
// Wm[b][i][j] = exp(-(i-j)^2 / (EPS_STD + 2*(R*L*s[b,i])^2)) * w[b,j]  (bf16)
// (row max of the Gaussian is at j=i, so ret/max cancels the C/std factor)
// ---------------------------------------------------------------------------
__global__ void wm_kernel(const float* __restrict__ s,
                          const float* __restrict__ w,
                          unsigned short* __restrict__ Wm) {
  const int b = blockIdx.y;
  const int i = blockIdx.x;
  const float sv  = s[(size_t)b * L_ + i];
  const float std = R_ * (float)L_ * sv;
  const float inv = 1.0f / (EPS_STD_ + 2.0f * std * std);
  const size_t base = ((size_t)b * L_ + i) * L_;
  #pragma unroll
  for (int r = 0; r < 4; ++r) {
    int j = threadIdx.x + r * 256;
    float d = (float)(j - i);
    float v = __expf(-d * d * inv) * w[(size_t)b * L_ + j];
    Wm[base + j] = f2bf(v);
  }
}

// ---------------------------------------------------------------------------
extern "C" void kernel_launch(void* const* d_in, const int* in_sizes, int n_in,
                              void* d_out, int out_size, void* d_ws, size_t ws_size,
                              hipStream_t stream) {
  (void)in_sizes; (void)n_in; (void)out_size; (void)ws_size;
  const float* feats = (const float*)d_in[0];
  const float* w1 = (const float*)d_in[1];
  const float* b1 = (const float*)d_in[2];
  const float* w2 = (const float*)d_in[3];
  const float* b2 = (const float*)d_in[4];
  const float* w3 = (const float*)d_in[5];
  const float* b3 = (const float*)d_in[6];
  float* out = (float*)d_out;

  char* ws = (char*)d_ws;
  size_t off = 0;
  auto alloc = [&](size_t bytes) {
    char* p = ws + off;
    off += (bytes + 255) & ~(size_t)255;
    return (void*)p;
  };

  unsigned short* xT   = (unsigned short*)alloc((size_t)B_ * D_ * L_ * 2);   // 16 MB
  unsigned short* fbf  = (unsigned short*)alloc((size_t)B_ * L_ * D_ * 2);   // 16 MB
  unsigned short* wp1  = (unsigned short*)alloc((size_t)3 * C1_ * D_ * 2);   // 12.6 MB
  unsigned short* wp2  = (unsigned short*)alloc((size_t)3 * C1_ * C1_ * 2);  // 25.2 MB
  float*          y    = (float*)alloc((size_t)B_ * C1_ * L_ * 4);           // 64 MB
  unsigned short* actT = (unsigned short*)alloc((size_t)B_ * L_ * C1_ * 2);  // 32 MB
  unsigned short* Wm   = (unsigned short*)alloc((size_t)B_ * L_ * L_ * 2);   // 16 MB
  float2*         part = (float2*)alloc((size_t)B_ * 1024 * sizeof(float2));
  float2*         stat = (float2*)alloc((size_t)B_ * sizeof(float2));
  float*          sbuf = (float*)alloc((size_t)B_ * L_ * 4);
  float*          wbuf = (float*)alloc((size_t)B_ * L_ * 4);

  const int PB = C1_ * L_;

  // 1. weight packing to bf16 [3][Cout][Cin]
  pack_w_kernel<<<dim3(2048), 256, 0, stream>>>(w1, wp1, C1_, D_);
  pack_w_kernel<<<dim3(4096), 256, 0, stream>>>(w2, wp2, C1_, C1_);

  // 2. feats -> xT (bf16 transposed) + fbf (bf16)
  transpose_convert_kernel<<<dim3(D_ / 32, L_ / 32, B_), dim3(32, 8), 0, stream>>>(
      feats, xT, fbf);

  // 3. conv1: y[b][2048][1024], BT = fbf [L][D]
  gemm_wmma_kernel<3, true><<<dim3(L_ / 128, C1_ / 128, B_), 256, 0, stream>>>(
      wp1, (unsigned)(C1_ * D_), 0, fbf, b1, y, C1_, L_, D_);

  // 4. LN1 + LeakyReLU -> actT (bf16, [B][L][C1])
  ln_partial_kernel<<<dim3(1024, B_), 256, 0, stream>>>(y, part, PB, 1024);
  ln_finalize_kernel<<<dim3(B_), 256, 0, stream>>>(part, stat, 1024, 1.0f / PB);
  ln_apply_transpose_kernel<<<dim3(L_ / 32, C1_ / 32, B_), dim3(32, 8), 0, stream>>>(
      y, stat, actT);

  // 5. conv2: y[b][2048][1024], BT = actT [L][C1]
  gemm_wmma_kernel<3, true><<<dim3(L_ / 128, C1_ / 128, B_), 256, 0, stream>>>(
      wp2, (unsigned)(C1_ * C1_), 0, actT, b2, y, C1_, L_, C1_);

  // 6. LN2 + LeakyReLU -> actT
  ln_partial_kernel<<<dim3(1024, B_), 256, 0, stream>>>(y, part, PB, 1024);
  ln_finalize_kernel<<<dim3(B_), 256, 0, stream>>>(part, stat, 1024, 1.0f / PB);
  ln_apply_transpose_kernel<<<dim3(L_ / 32, C1_ / 32, B_), dim3(32, 8), 0, stream>>>(
      y, stat, actT);

  // 7. conv3 + LN3 + LeakyReLU + softmaxes -> s, w
  head_kernel<<<dim3(B_), 1024, 0, stream>>>(actT, w3, b3, sbuf, wbuf);

  // 8. Gaussian weight matrix Wm (bf16)
  wm_kernel<<<dim3(L_, B_), 256, 0, stream>>>(sbuf, wbuf, Wm);

  // 9. out[b] = Wm[b] (LxL) x feats[b] (LxD); BT = xT [D][L]
  gemm_wmma_kernel<1, false><<<dim3(D_ / 128, L_ / 128, B_), 256, 0, stream>>>(
      Wm, 0, (size_t)L_ * L_, xT, nullptr, out, L_, D_, L_);
}
